// CB_Attention_8787503087809
// MI455X (gfx1250) — matrix operand order repaired and checked by reading, not compiled, not run on hardware
//
#include <hip/hip_runtime.h>

typedef __attribute__((ext_vector_type(16))) _Float16 v16h;
typedef __attribute__((ext_vector_type(8)))  _Float16 v8h;
typedef __attribute__((ext_vector_type(8)))  float    v8f;

#define Bn 32
#define Hn 128
#define Sn 8192
#define WCOLS 384       // 3*H columns of W

// ---------------------------------------------------------------------------
// Branchless tanh: exact at saturation (e->0 => -1, e->inf => +1), no exec
// manipulation, so the WMMA region keeps EXEC all-ones without save/restore.
// ---------------------------------------------------------------------------
__device__ __forceinline__ float fast_tanh(float x) {
    float e = __expf(2.0f * x);
    return 1.0f - 2.0f * __builtin_amdgcn_rcpf(e + 1.0f);
}

// ---------------------------------------------------------------------------
// Kernel 0a: re-layout W[:, 0:256] into f16 A-fragments.
// Storage order (halves): [mt][kt][plane(2)][lane(32)][8] -> per fragment,
// each 16B "plane" is lane-major so a ds_load_b128 across 32 lanes reads 512
// contiguous LDS bytes (bank-conflict-free).
// WMMA 16-bit A 16x32 per-lane layout (ISA 7.12.2): lane l, half hh:
//   M = l%16,  K = (j<4 ? 2j : 16+2(j-4)) + 8*(l/16) + (hh&1),  j = hh>>1
// ---------------------------------------------------------------------------
__global__ __launch_bounds__(256) void prep_w(const float* __restrict__ W,
                                              _Float16* __restrict__ Af) {
    int idx  = blockIdx.x * 256 + threadIdx.x;   // 0 .. 32767 halves
    int h8   = idx & 7;
    int lane = (idx >> 3) & 31;
    int plane= (idx >> 8) & 1;
    int kt   = (idx >> 9) & 7;
    int mt   = idx >> 12;
    int hh = plane * 8 + h8;
    int M = lane & 15;
    int j = hh >> 1, p = hh & 1;
    int K = (j < 4 ? 2 * j : 16 + 2 * (j - 4)) + ((lane >> 4) << 3) + p;
    Af[idx] = (_Float16)W[(mt * 16 + M) * WCOLS + kt * 32 + K];
}

// ---------------------------------------------------------------------------
// Kernel 0b: bias[b,h] = sum_k W[h, 256+k] * decoder_hidden[b,k]
// ---------------------------------------------------------------------------
__global__ __launch_bounds__(128) void prep_bias(const float* __restrict__ W,
                                                 const float* __restrict__ dec,
                                                 float* __restrict__ bias) {
    int b = blockIdx.x, h = threadIdx.x;
    const float* wr = W + h * WCOLS + 2 * Hn;
    const float* dr = dec + b * Hn;
    float s = 0.f;
    #pragma unroll 8
    for (int k = 0; k < Hn; ++k) s += wr[k] * dr[k];
    bias[b * Hn + h] = s;
}

// ---------------------------------------------------------------------------
// Kernel 1: scores[b,s] = sum_h v[h] * tanh( (W12 @ [Sh;Dh])[h,s] + bias[b,h] )
// 8 waves/block, 16 s-columns per wave. A-fragments double-buffered in LDS via
// async-to-LDS DMA (ASYNCcnt), prefetched one m-tile ahead; inside a tile the
// two ds_load_b128 of fragment kt+1 are issued before the kt WMMA so every
// dscnt wait is covered by matrix-pipe execution.
// Pipeline per mt:  wait(own async) -> barrier -> stage(mt+1) -> compute(mt)
// (single barrier: it also orders mt-1's reads of buf[(mt+1)&1] before the
//  new async writes into it).
// ---------------------------------------------------------------------------
__global__ __launch_bounds__(256) void attn_scores(
        const float* __restrict__ Sh, const float* __restrict__ Dh,
        const _Float16* __restrict__ Af, const float* __restrict__ bias,
        const float* __restrict__ vvec, float* __restrict__ out) {
    // 2 buffers x (one m-tile = 8 fragments x 512 halves = 8 KB)
    __shared__ __align__(16) _Float16 AfL[2][4096];

    const int tid  = threadIdx.x;
    const int b    = blockIdx.y;
    const int wave = tid >> 5;
    const int lane = tid & 31;
    const int ncol = blockIdx.x * 128 + wave * 16 + (lane & 15);
    const int khalf = (lane >> 4) << 3;        // B: K = hh + 8*(lane/16) + (hh>=8 ? 8 : 0)
    const int mrow  = (lane >> 4) << 3;        // C: M = i  +  8*(lane/16)

    // -------- async stage of one 8 KB m-tile: 256 threads x 32 B ----------
    auto stage = [&](int mt, int buf) {
        unsigned lds0 = (unsigned)(size_t)(void*)&AfL[buf][tid * 16]; // bytes
        unsigned lds1 = lds0 + 16;
        const char* g = (const char*)(Af + (size_t)mt * 4096) + tid * 32;
        asm volatile(
            "global_load_async_to_lds_b128 %0, %2, off\n\t"
            "global_load_async_to_lds_b128 %1, %2, off offset:16"
            :: "v"(lds0), "v"(lds1), "v"(g) : "memory");
    };

    stage(0, 0);   // prologue: m-tile 0 in flight while B streams from HBM

    // -------- load all 8 B-fragments (coalesced along s, cvt f32->f16) ----
    const float* __restrict__ srcS = Sh + (size_t)b * Hn * Sn + ncol;
    const float* __restrict__ srcD = Dh + (size_t)b * Hn * Sn + ncol;
    v16h Bf[8];
    #pragma unroll
    for (int kt = 0; kt < 8; ++kt) {
        #pragma unroll
        for (int hh = 0; hh < 16; ++hh) {
            int k = kt * 32 + khalf + hh + (hh >= 8 ? 8 : 0); // 0..255
            float x = (k < Hn) ? srcS[(size_t)k * Sn]
                               : srcD[(size_t)(k - Hn) * Sn];
            Bf[kt][hh] = (_Float16)x;
        }
    }

    float partial = 0.f;
    #pragma unroll
    for (int mt = 0; mt < 8; ++mt) {
        // own stage for buf[mt&1] (issued last iteration) has landed:
        asm volatile("s_wait_asynccnt 0x0" ::: "memory");
        __syncthreads();          // all waves' stages visible; old reads done
        if (mt < 7)
            stage(mt + 1, (mt + 1) & 1);   // prefetch next tile during compute

        v8f acc;
        #pragma unroll
        for (int i = 0; i < 8; ++i)
            acc[i] = bias[b * Hn + mt * 16 + mrow + i];

        const _Float16* Ab = &AfL[mt & 1][0];
        v8h lo = *(const v8h*)(Ab +       lane * 8);
        v8h hi = *(const v8h*)(Ab + 256 + lane * 8);
        #pragma unroll
        for (int kt = 0; kt < 8; ++kt) {
            v8h lo_n = lo, hi_n = hi;
            if (kt < 7) {   // issue kt+1's LDS loads before the kt WMMA
                lo_n = *(const v8h*)(Ab + (kt + 1) * 512 +       lane * 8);
                hi_n = *(const v8h*)(Ab + (kt + 1) * 512 + 256 + lane * 8);
            }
            v16h Afrag = __builtin_shufflevector(lo, hi,
                0, 1, 2, 3, 4, 5, 6, 7, 8, 9, 10, 11, 12, 13, 14, 15);
            acc = __builtin_amdgcn_wmma_f32_16x16x32_f16(
                      false, Afrag, false, Bf[kt], (short)0, acc, false, false);
            lo = lo_n; hi = hi_n;
        }

        #pragma unroll
        for (int i = 0; i < 8; ++i)
            partial += vvec[mt * 16 + mrow + i] * fast_tanh(acc[i]);
    }

    // lanes l and l^16 hold complementary M-halves of the same column
    partial += __shfl_xor(partial, 16, 32);
    if ((lane & 16) == 0)
        out[(size_t)b * Sn + ncol] = partial;
}

// ---------------------------------------------------------------------------
// Kernel 2: in-place softmax over each batch row of 8192 scores.
// ---------------------------------------------------------------------------
__global__ __launch_bounds__(256) void softmax_rows(float* __restrict__ out) {
    const int b = blockIdx.x;
    float* row = out + (size_t)b * Sn;
    __shared__ float red[8];

    float vals[32];
    float m = -3.402823466e38f;
    #pragma unroll
    for (int i = 0; i < 32; ++i) {
        vals[i] = row[threadIdx.x + i * 256];
        m = fmaxf(m, vals[i]);
    }
    #pragma unroll
    for (int off = 16; off > 0; off >>= 1)
        m = fmaxf(m, __shfl_xor(m, off, 32));
    if ((threadIdx.x & 31) == 0) red[threadIdx.x >> 5] = m;
    __syncthreads();
    float bm = red[0];
    #pragma unroll
    for (int i = 1; i < 8; ++i) bm = fmaxf(bm, red[i]);
    __syncthreads();

    float s = 0.f;
    #pragma unroll
    for (int i = 0; i < 32; ++i) {
        float e = __expf(vals[i] - bm);
        vals[i] = e;
        s += e;
    }
    #pragma unroll
    for (int off = 16; off > 0; off >>= 1)
        s += __shfl_xor(s, off, 32);
    if ((threadIdx.x & 31) == 0) red[threadIdx.x >> 5] = s;
    __syncthreads();
    float bs = 0.f;
    #pragma unroll
    for (int i = 0; i < 8; ++i) bs += red[i];
    float inv = 1.0f / bs;

    #pragma unroll
    for (int i = 0; i < 32; ++i)
        row[threadIdx.x + i * 256] = vals[i] * inv;
}

// ---------------------------------------------------------------------------
extern "C" void kernel_launch(void* const* d_in, const int* in_sizes, int n_in,
                              void* d_out, int out_size, void* d_ws, size_t ws_size,
                              hipStream_t stream) {
    const float* Sh  = (const float*)d_in[0];   // static_hidden  [32,128,8192]
    const float* Dh  = (const float*)d_in[1];   // dynamic_hidden [32,128,8192]
    const float* dec = (const float*)d_in[2];   // decoder_hidden [32,128]
    const float* v   = (const float*)d_in[3];   // v [1,1,128]
    const float* W   = (const float*)d_in[4];   // W [1,128,384]
    float* out = (float*)d_out;                 // [32,1,8192]

    _Float16* Af = (_Float16*)d_ws;                                   // 64 KB
    float* bias  = (float*)((char*)d_ws + 32768 * sizeof(_Float16));  // 16 KB

    prep_w   <<<128, 256, 0, stream>>>(W, Af);
    prep_bias<<<Bn, Hn, 0, stream>>>(W, dec, bias);

    dim3 grid(Sn / 128, Bn);   // 128 s-columns per block, one batch per y
    attn_scores<<<grid, 256, 0, stream>>>(Sh, Dh, Af, bias, v, out);

    softmax_rows<<<Bn, 256, 0, stream>>>(out);
}
